// HebbianMemory_46978352284396
// MI455X (gfx1250) — matrix operand order repaired and checked by reading, not compiled, not run on hardware
//
#include <hip/hip_runtime.h>
#include <hip/hip_bf16.h>

typedef __attribute__((ext_vector_type(16))) __bf16 v16bf;
typedef __attribute__((ext_vector_type(8)))  float  v8f;
typedef __attribute__((ext_vector_type(4)))  unsigned int v4u;
typedef __attribute__((ext_vector_type(8)))  int   v8i;
typedef __attribute__((ext_vector_type(4)))  int   v4i;

#define DM   512
#define HN   8
#define EH   64
#define BBAT 2
#define LSEQ 2048
#define CHNK 64

// ---------- bf16 helpers (RNE) ----------
__device__ __forceinline__ unsigned short f2bf(float f) {
  unsigned int x = __float_as_uint(f);
  x += 0x7fffu + ((x >> 16) & 1u);
  return (unsigned short)(x >> 16);
}
__device__ __forceinline__ float bf2f(unsigned short u) {
  return __uint_as_float(((unsigned int)u) << 16);
}

// ---------- WMMA operand loaders ----------
__device__ __forceinline__ v16bf load_a16(const unsigned short* base, int lane, int stride) {
  const unsigned short* p = base + (lane & 15) * stride + ((lane >> 4) << 3);
  union { uint4 u[2]; v16bf v; } r;
  r.u[0] = *reinterpret_cast<const uint4*>(p);
  r.u[1] = *reinterpret_cast<const uint4*>(p + 16);
  return r.v;
}
__device__ __forceinline__ v16bf load_b16(const unsigned short* base, int lane, int stride) {
  const unsigned short* p = base + (lane & 15) * stride + ((lane >> 4) << 3);
  union { uint4 u[2]; v16bf v; } r;
  r.u[0] = *reinterpret_cast<const uint4*>(p);
  r.u[1] = *reinterpret_cast<const uint4*>(p + 16);
  return r.v;
}
__device__ __forceinline__ v8f wmma_bf16(v16bf a, v16bf b, v8f c) {
  return __builtin_amdgcn_wmma_f32_16x16x32_bf16(false, a, false, b, (short)0, c, false, false);
}

// ---------- TDM: async 3D tile load  global -> LDS ----------
// tile: x = 64 contiguous f32 (one head slice), y = 3 (q,k,v at stride 512 elems),
//       z = NROW rows at stride 1536 elems.  LDS layout: stage[row][sel][64].
__device__ __forceinline__ void tdm_load_qkv3d(const float* gptr, unsigned lds_byte_off, int nrow) {
  unsigned long long ga = (unsigned long long)(size_t)gptr;
  v4u g0;
  g0.x = 1u;                                             // count=1, user mode
  g0.y = lds_byte_off;                                   // lds_addr (bytes)
  g0.z = (unsigned)(ga & 0xffffffffu);                   // global_addr lo
  g0.w = (unsigned)((ga >> 32) & 0x1ffffffu) | (2u << 30); // global_addr hi | type=2
  v8i g1;
  g1[0] = (2 << 16);                                     // workgroup_mask=0, data_size=4B
  g1[1] = (64 << 16);                                    // tensor_dim0 = 64   (bits 79:48)
  g1[2] = (3 << 16);                                     // tensor_dim1 = 3    (bits 111:80)
  g1[3] = (64 << 16);                                    // tile_dim0 = 64     (bits 127:112)
  g1[4] = 3 | (nrow << 16);                              // tile_dim1=3, tile_dim2=nrow
  g1[5] = 512;                                           // tensor_dim0_stride = 512 elems
  g1[6] = (1536 << 16);                                  // tensor_dim1_stride = 1536 elems
  g1[7] = 0;
  v4i g2 = {64, 0, 0, 0};                                // tensor_dim2 = 64
  v4i g3 = {0, 0, 0, 0};
#if __clang_major__ >= 23
  v8i g4 = {0, 0, 0, 0, 0, 0, 0, 0};
  __builtin_amdgcn_tensor_load_to_lds(g0, g1, g2, g3, g4, 0);
#else
  __builtin_amdgcn_tensor_load_to_lds(g0, g1, g2, g3, 0);
#endif
}

// ================= GEMM: Out = A[M,K] @ B[K,N] + bias (+ QKV feature map) =================
__global__ __launch_bounds__(128) void gemm_bf16_wmma(
    const float* __restrict__ A, const float* __restrict__ Bm,
    const float* __restrict__ bias, float* __restrict__ Out,
    int M, int K, int N, int mode)
{
  __shared__ __align__(16) unsigned short As[64][40];
  __shared__ __align__(16) unsigned short Bt[64][40];
  const int tid = threadIdx.x, wave = tid >> 5, lane = tid & 31;
  const int m0 = blockIdx.y * 64, n0 = blockIdx.x * 64;

  v8f acc[4] = {v8f{}, v8f{}, v8f{}, v8f{}};

  for (int k0 = 0; k0 < K; k0 += 32) {
    __syncthreads();
    for (int o = tid; o < 64 * 32; o += 128) {
      int i = o >> 5, k = o & 31;
      As[i][k] = f2bf(A[(size_t)(m0 + i) * K + (k0 + k)]);
    }
    for (int o = tid; o < 32 * 64; o += 128) {
      int k = o >> 6, n = o & 63;
      Bt[n][k] = f2bf(Bm[(size_t)(k0 + k) * N + (n0 + n)]);
    }
    if (k0 + 32 < K) {   // prefetch next K-slab (global_prefetch_b8)
      __builtin_prefetch(&A[(size_t)(m0 + (tid >> 1)) * K + (k0 + 32) + (tid & 1) * 16], 0, 0);
      __builtin_prefetch(&Bm[(size_t)(k0 + 32 + (tid >> 2)) * N + n0 + (tid & 3) * 16], 0, 0);
    }
    __syncthreads();
    v16bf av = load_a16(&As[wave * 16][0], lane, 40);
    for (int t = 0; t < 4; ++t) {
      v16bf bv = load_b16(&Bt[t * 16][0], lane, 40);
      acc[t] = wmma_bf16(av, bv, acc[t]);
    }
  }

  for (int t = 0; t < 4; ++t)
    for (int r = 0; r < 8; ++r) {
      int m = m0 + wave * 16 + r + ((lane >> 4) << 3);
      int n = n0 + t * 16 + (lane & 15);
      float v = acc[t][r] + bias[n];
      if (mode == 1 && n < 2 * DM) {
        float e = (v > 0.f) ? (v + 1.f) : __expf(v);
        v = (n < DM) ? e * 0.125f : e;
      }
      Out[(size_t)m * N + n] = v;
    }
}

// ================= Chunked decayed linear attention =================
__global__ __launch_bounds__(128) void hebbian_attn(
    const float* __restrict__ qkv, const float* __restrict__ decay_logits,
    float* __restrict__ attn)
{
  __shared__ __align__(16) unsigned short qd [64][72]; // q_i * lam^i
  __shared__ __align__(16) unsigned short kd [64][72]; // k_j * lam^-j
  __shared__ __align__(16) unsigned short kdT[64][72];
  __shared__ __align__(16) unsigned short vt [64][72]; // v^T
  __shared__ __align__(16) unsigned short st [64][72]; // S_prev^T (bf16)
  __shared__ __align__(16) union Ovl {                 // am dead while staging live
    unsigned short am[64][72];                         // masked A (bf16)
    float stage[16][3][64];                            // TDM staging (fp32)
  } ovl;
  __shared__ float zf[64], ksum[64], den[64];

  const int tid = threadIdx.x, wave = tid >> 5, lane = tid & 31;
  const int b = blockIdx.x / HN, h = blockIdx.x % HN;
  const float lam  = 1.f / (1.f + __expf(-decay_logits[h]));
  const float l2l  = __log2f(lam);
  const float lamC  = exp2f(l2l * (float)CHNK);
  const float lamC1 = exp2f(l2l * (float)(CHNK - 1));
  const unsigned stage_off = (unsigned)(size_t)(void*)&ovl.stage[0][0][0];

  for (int o = tid; o < 64 * 72; o += 128) (&st[0][0])[o] = 0;
  if (tid < 64) zf[tid] = 0.f;

  v8f Sacc[4] = {v8f{}, v8f{}, v8f{}, v8f{}};

  for (int c = 0; c < LSEQ / CHNK; ++c) {
    __syncthreads();
    const size_t rowbase = (size_t)b * LSEQ + (size_t)c * CHNK;
    // ---- P0: TDM-load chunk in 4 slabs of 16 rows, convert with local decay ----
    for (int rr = 0; rr < 4; ++rr) {
      if (wave == 0) {
        tdm_load_qkv3d(qkv + (rowbase + rr * 16) * (3 * DM) + h * EH, stage_off, 16);
        __builtin_amdgcn_s_wait_tensorcnt(0);
      }
      __syncthreads();
      for (int o = tid; o < 16 * 64; o += 128) {
        int il = o >> 6, e = o & 63;
        int i = rr * 16 + il;
        float q = ovl.stage[il][0][e], k = ovl.stage[il][1][e], v = ovl.stage[il][2][e];
        float dq = exp2f(l2l * (float)i);
        unsigned short kb = f2bf(k * exp2f(-l2l * (float)i));
        qd[i][e] = f2bf(q * dq);
        kd[i][e] = kb;
        kdT[e][i] = kb;
        vt[e][i] = f2bf(v);
      }
      __syncthreads();
    }
    // ---- P0b: den init + kd column sums ----
    if (tid < 64) {
      float s = 0.f;
      for (int e = 0; e < 64; ++e) s += bf2f(qd[tid][e]) * zf[e];
      den[tid] = lam * s;
      float ks = 0.f;
      for (int j = 0; j < 64; ++j) ks += bf2f(kd[j][tid]);
      ksum[tid] = ks;
    }
    __syncthreads();
    // ---- P1: A = Qd Kd^T, causal mask, row-sum into den ----
    v16bf aq0 = load_a16(&qd[wave * 16][0],  lane, 72);
    v16bf aq1 = load_a16(&qd[wave * 16][32], lane, 72);
    for (int t = 0; t < 4; ++t) {
      v8f a0 = v8f{};
      a0 = wmma_bf16(aq0, load_b16(&kd[t * 16][0],  lane, 72), a0);
      a0 = wmma_bf16(aq1, load_b16(&kd[t * 16][32], lane, 72), a0);
      for (int r = 0; r < 8; ++r) {
        int m = wave * 16 + r + ((lane >> 4) << 3);
        int j = t * 16 + (lane & 15);
        float mv = (j <= m) ? a0[r] : 0.f;
        ovl.am[m][j] = f2bf(mv);
        float red = mv;
        for (int off = 1; off < 16; off <<= 1) red += __shfl_xor(red, off, 16);
        if ((lane & 15) == 0) atomicAdd(&den[m], red);
      }
    }
    __syncthreads();
    // ---- P2: num = lam * Qd@S_prev + Amask@V ; out = num/(den+eps) ----
    v16bf am0 = load_a16(&ovl.am[wave * 16][0],  lane, 72);
    v16bf am1 = load_a16(&ovl.am[wave * 16][32], lane, 72);
    for (int t = 0; t < 4; ++t) {
      v8f acc = v8f{};
      acc = wmma_bf16(aq0, load_b16(&st[t * 16][0],  lane, 72), acc);
      acc = wmma_bf16(aq1, load_b16(&st[t * 16][32], lane, 72), acc);
      for (int r = 0; r < 8; ++r) acc[r] *= lam;
      acc = wmma_bf16(am0, load_b16(&vt[t * 16][0],  lane, 72), acc);
      acc = wmma_bf16(am1, load_b16(&vt[t * 16][32], lane, 72), acc);
      for (int r = 0; r < 8; ++r) {
        int i = wave * 16 + r + ((lane >> 4) << 3);
        int f = t * 16 + (lane & 15);
        attn[(rowbase + i) * DM + h * EH + f] = acc[r] / (den[i] + 1e-5f);
      }
    }
    __syncthreads();
    // ---- P3: S = lam^C * S + lam^(C-1) * Kd^T V ; z likewise ----
    v16bf ak0 = load_a16(&kdT[wave * 16][0],  lane, 72);
    v16bf ak1 = load_a16(&kdT[wave * 16][32], lane, 72);
    for (int t = 0; t < 4; ++t) {
      v8f T = v8f{};
      T = wmma_bf16(ak0, load_b16(&vt[t * 16][0],  lane, 72), T);
      T = wmma_bf16(ak1, load_b16(&vt[t * 16][32], lane, 72), T);
      for (int r = 0; r < 8; ++r) {
        Sacc[t][r] = Sacc[t][r] * lamC + T[r] * lamC1;
        int e = wave * 16 + r + ((lane >> 4) << 3);
        int f = t * 16 + (lane & 15);
        st[f][e] = f2bf(Sacc[t][r]);
      }
    }
    if (tid < 64) zf[tid] = zf[tid] * lamC + ksum[tid] * lamC1;
  }
}

// ================= LayerNorm over last dim (512) =================
__global__ __launch_bounds__(256) void layernorm512(
    const float* __restrict__ X, const float* __restrict__ gamma,
    const float* __restrict__ beta, float* __restrict__ out)
{
  __shared__ float red[256];
  const int tid = threadIdx.x;
  const size_t row = blockIdx.x;
  const float* xr = X + row * DM;
  float v0 = xr[tid], v1 = xr[tid + 256];
  red[tid] = v0 + v1;
  __syncthreads();
  for (int o = 128; o > 0; o >>= 1) { if (tid < o) red[tid] += red[tid + o]; __syncthreads(); }
  float mu = red[0] * (1.f / DM);
  __syncthreads();
  float d0 = v0 - mu, d1 = v1 - mu;
  red[tid] = d0 * d0 + d1 * d1;
  __syncthreads();
  for (int o = 128; o > 0; o >>= 1) { if (tid < o) red[tid] += red[tid + o]; __syncthreads(); }
  float rstd = rsqrtf(red[0] * (1.f / DM) + 1e-5f);
  out[row * DM + tid]       = d0 * rstd * gamma[tid]       + beta[tid];
  out[row * DM + tid + 256] = d1 * rstd * gamma[tid + 256] + beta[tid + 256];
}

// ================= launch =================
extern "C" void kernel_launch(void* const* d_in, const int* in_sizes, int n_in,
                              void* d_out, int out_size, void* d_ws, size_t ws_size,
                              hipStream_t stream) {
  const float* x     = (const float*)d_in[0];
  const float* Wqkv  = (const float*)d_in[1];
  const float* bqkv  = (const float*)d_in[2];
  const float* Wout  = (const float*)d_in[3];
  const float* bout  = (const float*)d_in[4];
  const float* gamma = (const float*)d_in[5];
  const float* beta  = (const float*)d_in[6];
  const float* dlog  = (const float*)d_in[7];
  float* out = (float*)d_out;

  const int M = BBAT * LSEQ;
  float* qkv  = (float*)d_ws;
  float* attn = qkv  + (size_t)M * 3 * DM;
  float* proj = attn + (size_t)M * DM;

  gemm_bf16_wmma<<<dim3(3 * DM / 64, M / 64), dim3(128), 0, stream>>>(
      x, Wqkv, bqkv, qkv, M, DM, 3 * DM, 1);
  hebbian_attn<<<dim3(BBAT * HN), dim3(128), 0, stream>>>(qkv, dlog, attn);
  gemm_bf16_wmma<<<dim3(DM / 64, M / 64), dim3(128), 0, stream>>>(
      attn, Wout, bout, proj, M, DM, DM, 0);
  layernorm512<<<dim3(M), dim3(256), 0, stream>>>(proj, gamma, beta, out);
}
